// CIAttention_2654289789117
// MI455X (gfx1250) — matrix operand order repaired and checked by reading, compiled, not run on hardware
//
#include <hip/hip_runtime.h>

// ---------------------------------------------------------------------------
// CDNA5 (gfx1250, wave32) fused MHA: QKV proj (bf16 WMMA, double-buffered
// LDS) + RoPE + flash attention (bf16 WMMA, fp32 online softmax, LDS-staged
// K/V with async global->LDS) + output proj.
// ---------------------------------------------------------------------------

typedef __attribute__((ext_vector_type(16))) __bf16 bf16x16;
typedef __attribute__((ext_vector_type(8)))  __bf16 bf16x8;
typedef __attribute__((ext_vector_type(4)))  __bf16 bf16x4;
typedef __attribute__((ext_vector_type(8)))  float  f32x8;
typedef int v4i __attribute__((vector_size(16)));

#define D_MODEL 2048
#define N_HEADS 16
#define HEAD_DIM 128
#define BATCH 4
#define SEQ 2048
#define TOKENS (BATCH * SEQ)   // 8192

#if defined(__gfx1250__) && __has_builtin(__builtin_amdgcn_global_load_async_to_lds_b128) && __has_builtin(__builtin_amdgcn_s_wait_asynccnt)
#define USE_ASYNC_LDS 1
#else
#define USE_ASYNC_LDS 0
#endif

__device__ __forceinline__ __bf16 f2b(float f) { return (__bf16)f; }

__device__ __forceinline__ f32x8 zero_f32x8() {
    f32x8 z;
#pragma unroll
    for (int i = 0; i < 8; ++i) z[i] = 0.0f;
    return z;
}

#if USE_ASYNC_LDS
// Prototype (from clang-22 diagnostic): (v4i __device__*, v4i __shared__*, imm, imm)
__device__ __forceinline__ void async_copy16(__bf16* lds, const __bf16* g) {
    __builtin_amdgcn_global_load_async_to_lds_b128(
        (__attribute__((address_space(1))) v4i*)(void*)g,
        (__attribute__((address_space(3))) v4i*)(void*)lds, 0, 0);
}
#endif

// A operand, 16x32 bf16 (MxK): lane holds row M=lane&15; K chunks at
// g*8 and 16+g*8 where g=lane>>4 (CDNA5 ISA 7.12.2). Two 16B loads.
__device__ __forceinline__ bf16x16 load_frag_a(const __bf16* base, int ld) {
    const int lane = threadIdx.x & 31;
    const int row  = lane & 15;
    const int g    = lane >> 4;
    const __bf16* p = base + row * ld + g * 8;
    bf16x8 lo = *reinterpret_cast<const bf16x8*>(p);
    bf16x8 hi = *reinterpret_cast<const bf16x8*>(p + 16);
    bf16x16 r;
#pragma unroll
    for (int i = 0; i < 8; ++i) { r[i] = lo[i]; r[i + 8] = hi[i]; }
    return r;
}

// B operand, 32x16 bf16 (KxN): lane holds col N=lane&15; K chunks at
// g*8 and 16+g*8 where g=lane>>4. Source is an NxK row-major tile.
__device__ __forceinline__ bf16x16 load_frag_b(const __bf16* base, long ld) {
    const int lane = threadIdx.x & 31;
    const int n    = lane & 15;
    const int g    = lane >> 4;
    const __bf16* p = base + (long)n * ld + g * 8;
    bf16x8 lo = *reinterpret_cast<const bf16x8*>(p);
    bf16x8 hi = *reinterpret_cast<const bf16x8*>(p + 16);
    bf16x16 r;
#pragma unroll
    for (int i = 0; i < 8; ++i) { r[i] = lo[i]; r[i + 8] = hi[i]; }
    return r;
}

__device__ __forceinline__ f32x8 wmma_bf16(bf16x16 a, bf16x16 b, f32x8 c) {
    return __builtin_amdgcn_wmma_f32_16x16x32_bf16(
        false, a, false, b, (short)0, c, false, false);
}

// ---------------------------------------------------------------------------
// Projection GEMM: C[M=8192, N=2048] = A[M,K=2048] * W[N,K]^T.
// mode 0/1: A=x fp32 -> bf16, dst = q/k  as [B,H,S,Dh] bf16
// mode 2  : A=x fp32 -> bf16, dst = v^T  as [B,H,Dh,S] bf16
// mode 3  : A=attn_out bf16,  dst = fp32 [TOKENS, D_MODEL]
// 256 threads (8 waves), tile 128x128, BK=32, double-buffered LDS (pad 40).
// ---------------------------------------------------------------------------
__global__ __launch_bounds__(256) void proj_gemm(
    const float* __restrict__ A32, const __bf16* __restrict__ A16,
    const float* __restrict__ W, void* __restrict__ dst, int mode) {

    const int a0 = blockIdx.x * 128;
    const int n0 = blockIdx.y * 128;
    const int tid  = threadIdx.x;
    const int lane = tid & 31;
    const int w    = tid >> 5;
    const int wm   = w >> 2;   // 0..1
    const int wn   = w & 3;    // 0..3
    const int g    = lane >> 4;
    const int ln   = lane & 15;

    __shared__ __bf16 As[2][128 * 40];
    __shared__ __bf16 Bs[2][128 * 40];

    float4 ra[4], rb[4];
    bf16x8 ra16[2];

    // Issue ALL global loads for one K-tile into registers (no interleaved
    // waits); conversion + LDS stores happen later in store_tiles.
    auto load_tiles = [&](int k0) {
        if (mode == 3) {
#pragma unroll
            for (int i = 0; i < 2; ++i) {
                const int c = tid + i * 256;
                const int row = c >> 2, col = (c & 3) * 8;
                ra16[i] = *reinterpret_cast<const bf16x8*>(
                    A16 + (size_t)(a0 + row) * D_MODEL + k0 + col);
            }
        } else {
#pragma unroll
            for (int i = 0; i < 4; ++i) {
                const int c = tid + i * 256;
                const int row = c >> 3, col = (c & 7) * 4;
                ra[i] = *reinterpret_cast<const float4*>(
                    A32 + (size_t)(a0 + row) * D_MODEL + k0 + col);
            }
        }
#pragma unroll
        for (int i = 0; i < 4; ++i) {
            const int c = tid + i * 256;
            const int row = c >> 3, col = (c & 7) * 4;
            rb[i] = *reinterpret_cast<const float4*>(
                W + (size_t)(n0 + row) * D_MODEL + k0 + col);
        }
    };
    auto store_tiles = [&](int buf) {
        if (mode == 3) {
#pragma unroll
            for (int i = 0; i < 2; ++i) {
                const int c = tid + i * 256;
                const int row = c >> 2, col = (c & 3) * 8;
                *reinterpret_cast<bf16x8*>(&As[buf][row * 40 + col]) = ra16[i];
            }
        } else {
#pragma unroll
            for (int i = 0; i < 4; ++i) {
                const int c = tid + i * 256;
                const int row = c >> 3, col = (c & 7) * 4;
                bf16x4 b;
                b[0] = f2b(ra[i].x); b[1] = f2b(ra[i].y);
                b[2] = f2b(ra[i].z); b[3] = f2b(ra[i].w);
                *reinterpret_cast<bf16x4*>(&As[buf][row * 40 + col]) = b;
            }
        }
#pragma unroll
        for (int i = 0; i < 4; ++i) {
            const int c = tid + i * 256;
            const int row = c >> 3, col = (c & 7) * 4;
            bf16x4 b;
            b[0] = f2b(rb[i].x); b[1] = f2b(rb[i].y);
            b[2] = f2b(rb[i].z); b[3] = f2b(rb[i].w);
            *reinterpret_cast<bf16x4*>(&Bs[buf][row * 40 + col]) = b;
        }
    };

    f32x8 acc[4][2];
#pragma unroll
    for (int mi = 0; mi < 4; ++mi)
#pragma unroll
        for (int nj = 0; nj < 2; ++nj) acc[mi][nj] = zero_f32x8();

    load_tiles(0);
    store_tiles(0);
    __syncthreads();

    int cur = 0;
    for (int k0 = 0; k0 < D_MODEL; k0 += 32) {
        const bool more = (k0 + 32) < D_MODEL;
        if (more) load_tiles(k0 + 32);                 // overlap with WMMA
        if (k0 + 64 < D_MODEL) {                       // 2 tiles ahead
            const int prow = tid & 127;
            __builtin_prefetch(W + (size_t)(n0 + prow) * D_MODEL + k0 + 64, 0, 1);
            if (mode != 3)
                __builtin_prefetch(A32 + (size_t)(a0 + prow) * D_MODEL + k0 + 64, 0, 1);
        }

        bf16x16 af[4], bfr[2];
#pragma unroll
        for (int mi = 0; mi < 4; ++mi)
            af[mi] = load_frag_a(&As[cur][(wm * 64 + mi * 16) * 40], 40);
#pragma unroll
        for (int nj = 0; nj < 2; ++nj)
            bfr[nj] = load_frag_b(&Bs[cur][(wn * 32 + nj * 16) * 40], 40);
#pragma unroll
        for (int mi = 0; mi < 4; ++mi)
#pragma unroll
            for (int nj = 0; nj < 2; ++nj)
                acc[mi][nj] = wmma_bf16(af[mi], bfr[nj], acc[mi][nj]);

        if (more) store_tiles(cur ^ 1);
        __syncthreads();
        cur ^= 1;
    }

    // --- epilogue ---
#pragma unroll
    for (int mi = 0; mi < 4; ++mi) {
#pragma unroll
        for (int nj = 0; nj < 2; ++nj) {
#pragma unroll
            for (int r = 0; r < 8; ++r) {
                const int grow = a0 + wm * 64 + mi * 16 + r + 8 * g; // token
                const int gcol = n0 + wn * 32 + nj * 16 + ln;        // feature
                const float val = acc[mi][nj][r];
                if (mode == 3) {
                    ((float*)dst)[(size_t)grow * D_MODEL + gcol] = val;
                } else {
                    const int b  = grow >> 11;
                    const int s  = grow & (SEQ - 1);
                    const int h  = gcol >> 7;
                    const int dh = gcol & (HEAD_DIM - 1);
                    __bf16* o = (__bf16*)dst;
                    if (mode == 2) {
                        o[((size_t)(b * N_HEADS + h) * HEAD_DIM + dh) * SEQ + s] = f2b(val);
                    } else {
                        o[((size_t)(b * N_HEADS + h) * SEQ + s) * HEAD_DIM + dh] = f2b(val);
                    }
                }
            }
        }
    }
}

// ---------------------------------------------------------------------------
// RoPE in place on [B,H,S,Dh] bf16. One thread per (bh,s,j<64) pair.
// ---------------------------------------------------------------------------
__global__ __launch_bounds__(256) void rope_kernel(
    __bf16* __restrict__ t, const float* __restrict__ cosp,
    const float* __restrict__ sinp) {
    const size_t i = (size_t)blockIdx.x * 256 + threadIdx.x;
    const int j = (int)(i & 63);
    const int s = (int)((i >> 6) & (SEQ - 1));
    const size_t bh = i >> 17;
    const size_t base = (bh * SEQ + s) * HEAD_DIM + j;
    const float x1 = (float)t[base];
    const float x2 = (float)t[base + 64];
    const float c  = cosp[s * 64 + j];
    const float sn = sinp[s * 64 + j];
    t[base]      = f2b(x1 * c - x2 * sn);
    t[base + 64] = f2b(x1 * sn + x2 * c);
}

// ---------------------------------------------------------------------------
// Flash attention. 128 threads (4 waves); each wave owns 16 query rows.
// grid.x = B*H*32. Q,K: [B,H,S,Dh] bf16, V: [B,H,Dh,S] bf16.
// K/V blocks staged into LDS once per block (async global->LDS) and shared
// by all 4 waves. Output: [B,S,D_MODEL] bf16.
// ---------------------------------------------------------------------------
__global__ __launch_bounds__(128) void attn_kernel(
    const __bf16* __restrict__ q, const __bf16* __restrict__ k,
    const __bf16* __restrict__ v, __bf16* __restrict__ out) {

    const int tid  = threadIdx.x;
    const int lane = tid & 31;
    const int w    = tid >> 5;
    const int bh   = blockIdx.x >> 5;
    const int qblk = blockIdx.x & 31;
    const int b    = bh >> 4;
    const int h    = bh & 15;
    const int q0   = qblk * 64 + w * 16;
    const int g    = lane >> 4;
    const int ln   = lane & 15;
    const float scale = 0.08838834764831845f; // 1/sqrt(128)

    __shared__ __bf16 Ks[32 * 136];       // 32 keys x 128 dh, pad 136
    __shared__ __bf16 Vs[128 * 40];       // 128 dh x 32 keys, pad 40
    __shared__ __bf16 Pl[4][16 * 32];     // per-wave P transpose scratch
    __bf16* Pw = Pl[w];

    const __bf16* qbase = q + ((size_t)bh * SEQ + q0) * HEAD_DIM;
    bf16x16 qf[4];
#pragma unroll
    for (int kk = 0; kk < 4; ++kk)
        qf[kk] = load_frag_a(qbase + kk * 32, HEAD_DIM);

    f32x8 o[8];
#pragma unroll
    for (int j = 0; j < 8; ++j) o[j] = zero_f32x8();
    float m_r[8], l_r[8];
#pragma unroll
    for (int r = 0; r < 8; ++r) { m_r[r] = -1e30f; l_r[r] = 0.0f; }

    for (int kb = 0; kb < SEQ; kb += 32) {
        const __bf16* kbase = k + ((size_t)bh * SEQ + kb) * HEAD_DIM;
        const __bf16* vbase = v + (size_t)bh * HEAD_DIM * SEQ + kb;

        // ---- cooperative K/V staging (512 16B chunks each) ----
#if USE_ASYNC_LDS
#pragma unroll
        for (int i = 0; i < 4; ++i) {
            const int c = tid + i * 128;
            const int kr = c >> 4, kc = (c & 15) * 8;      // K: 32 x (16 chunks)
            async_copy16(&Ks[kr * 136 + kc], kbase + kr * HEAD_DIM + kc);
            const int vr = c >> 2, vc = (c & 3) * 8;       // V: 128 x (4 chunks)
            async_copy16(&Vs[vr * 40 + vc], vbase + (size_t)vr * SEQ + vc);
        }
        __builtin_amdgcn_s_wait_asynccnt(0);
#else
        bf16x8 tk[4], tv[4];
#pragma unroll
        for (int i = 0; i < 4; ++i) {
            const int c = tid + i * 128;
            tk[i] = *reinterpret_cast<const bf16x8*>(
                kbase + (c >> 4) * HEAD_DIM + (c & 15) * 8);
            tv[i] = *reinterpret_cast<const bf16x8*>(
                vbase + (size_t)(c >> 2) * SEQ + (c & 3) * 8);
        }
#pragma unroll
        for (int i = 0; i < 4; ++i) {
            const int c = tid + i * 128;
            *reinterpret_cast<bf16x8*>(&Ks[(c >> 4) * 136 + (c & 15) * 8]) = tk[i];
            *reinterpret_cast<bf16x8*>(&Vs[(c >> 2) * 40 + (c & 3) * 8]) = tv[i];
        }
#endif
        __syncthreads();

        // ---- scores: two 16x16 fragments from LDS K ----
        f32x8 sc[2];
#pragma unroll
        for (int cf = 0; cf < 2; ++cf) {
            f32x8 s = zero_f32x8();
#pragma unroll
            for (int kk = 0; kk < 4; ++kk)
                s = wmma_bf16(qf[kk],
                              load_frag_b(&Ks[(cf * 16) * 136 + kk * 32], 136),
                              s);
            sc[cf] = s;
        }

        // ---- online softmax (fp32, 16-lane group reductions) ----
        float alpha[8];
#pragma unroll
        for (int r = 0; r < 8; ++r) {
            const float s0 = sc[0][r] * scale;
            const float s1 = sc[1][r] * scale;
            float t = fmaxf(s0, s1);
#pragma unroll
            for (int mk = 1; mk < 16; mk <<= 1)
                t = fmaxf(t, __shfl_xor(t, mk));
            const float mn = fmaxf(m_r[r], t);
            alpha[r] = __expf(m_r[r] - mn);
            m_r[r] = mn;
            const float p0 = __expf(s0 - mn);
            const float p1 = __expf(s1 - mn);
            float ps = p0 + p1;
#pragma unroll
            for (int mk = 1; mk < 16; mk <<= 1)
                ps += __shfl_xor(ps, mk);
            l_r[r] = l_r[r] * alpha[r] + ps;
            const int rowM = r + 8 * g;
            Pw[rowM * 32 + ln]      = f2b(p0);
            Pw[rowM * 32 + ln + 16] = f2b(p1);
        }
#pragma unroll
        for (int j = 0; j < 8; ++j)
#pragma unroll
            for (int r = 0; r < 8; ++r) o[j][r] *= alpha[r];

        // ---- P through per-wave LDS -> A fragment; PV from LDS V ----
        const bf16x16 pf = load_frag_a(Pw, 32);
#pragma unroll
        for (int j = 0; j < 8; ++j)
            o[j] = wmma_bf16(pf, load_frag_b(&Vs[(j * 16) * 40], 40), o[j]);

        __syncthreads();   // protect Ks/Vs before next staging pass
    }

    // ---- epilogue: out[b, s, h*128+dh] = o / l ----
#pragma unroll
    for (int r = 0; r < 8; ++r) {
        const float inv = 1.0f / l_r[r];
        const int token = b * SEQ + q0 + r + 8 * g;
#pragma unroll
        for (int j = 0; j < 8; ++j) {
            const int col = h * HEAD_DIM + j * 16 + ln;
            out[(size_t)token * D_MODEL + col] = f2b(o[j][r] * inv);
        }
    }
}

// ---------------------------------------------------------------------------
extern "C" void kernel_launch(void* const* d_in, const int* in_sizes, int n_in,
                              void* d_out, int out_size, void* d_ws, size_t ws_size,
                              hipStream_t stream) {
    (void)in_sizes; (void)n_in; (void)out_size; (void)ws_size;
    const float* x    = (const float*)d_in[0];
    const float* cosp = (const float*)d_in[1];
    const float* sinp = (const float*)d_in[2];
    const float* Wq   = (const float*)d_in[3];
    const float* Wk   = (const float*)d_in[4];
    const float* Wv   = (const float*)d_in[5];
    const float* Wo   = (const float*)d_in[6];

    const size_t SZ = (size_t)BATCH * N_HEADS * SEQ * HEAD_DIM; // 16,777,216
    __bf16* q_ws = (__bf16*)d_ws;
    __bf16* k_ws = q_ws + SZ;
    __bf16* v_ws = k_ws + SZ;
    __bf16* ao   = v_ws + SZ;   // [TOKENS, D_MODEL] bf16

    dim3 gg(TOKENS / 128, D_MODEL / 128); // 64 x 16
    proj_gemm<<<gg, 256, 0, stream>>>(x, nullptr, Wq, q_ws, 0);
    proj_gemm<<<gg, 256, 0, stream>>>(x, nullptr, Wk, k_ws, 1);
    proj_gemm<<<gg, 256, 0, stream>>>(x, nullptr, Wv, v_ws, 2);

    const int rope_threads = BATCH * N_HEADS * SEQ * 64; // 8,388,608
    rope_kernel<<<rope_threads / 256, 256, 0, stream>>>(q_ws, cosp, sinp);
    rope_kernel<<<rope_threads / 256, 256, 0, stream>>>(k_ws, cosp, sinp);

    attn_kernel<<<BATCH * N_HEADS * 32, 128, 0, stream>>>(q_ws, k_ws, v_ws, ao);

    proj_gemm<<<gg, 256, 0, stream>>>(nullptr, ao, Wo, d_out, 3);
}
